// Speller_34574486733520
// MI455X (gfx1250) — matrix operand order repaired and checked by reading, not compiled, not run on hardware
//
#include <hip/hip_runtime.h>
#include <hip/hip_bf16.h>
#include <math.h>

// ---------------- problem constants ----------------
#define Bz   64
#define Lz   2048
#define Hz   512
#define Vz   34
#define Tz   600
#define H4   2048           // 4*H
#define KX   1536           // 2H + H  (LSTM fused K)

typedef __bf16 bf16_t;
typedef __attribute__((ext_vector_type(16))) __bf16 v16bf;
typedef __attribute__((ext_vector_type(8)))  float  v8f;
typedef unsigned int u32;

union V16BF { v16bf v; bf16_t e[16]; uint4 q[2]; };
union V8F   { v8f   v; float  e[8];  };

// ---------------- WMMA tile loaders (ISA §7.12.2 layouts, wave32) ----------------
// A 16x32 bf16, row-major source (row=M, K contiguous):
//   lane<16 : M=lane,    e[0..7]=K0..7,  e[8..15]=K16..23
//   lane>=16: M=lane-16, e[0..7]=K8..15, e[8..15]=K24..31
__device__ inline v16bf load_a_bf16(const bf16_t* base, int ld, int kbase, int lane) {
    int r  = lane & 15;
    int ko = (lane >> 4) << 3;          // 0 or 8
    const bf16_t* p = base + (size_t)r * ld + kbase + ko;
    V16BF u;
    u.q[0] = *(const uint4*)(p);
    u.q[1] = *(const uint4*)(p + 16);
    return u.v;
}
// same tile but from f32 source, converting in-register (for the one-shot K/V GEMM)
__device__ inline v16bf load_a_f32(const float* base, int ld, int kbase, int lane) {
    int r  = lane & 15;
    int ko = (lane >> 4) << 3;
    const float* p = base + (size_t)r * ld + kbase + ko;
    V16BF u;
#pragma unroll
    for (int j = 0; j < 8; ++j) {
        u.e[j]     = (bf16_t)p[j];
        u.e[8 + j] = (bf16_t)p[16 + j];
    }
    return u.v;
}
// B 32x16 bf16 built from row-major W[n][k] (so B(kk,n)=W[n][kbase+kk]):
//   lane<16 : N=lane,    e[0..7]=K0..7, e[8..15]=K16..23 ; lane>=16: N=lane-16, e[0..7]=K8..15, e[8..15]=K24..31
__device__ inline v16bf load_b_w(const bf16_t* W, int ldw, int n0, int kbase, int lane) {
    int n  = n0 + (lane & 15);
    int ko = (lane >> 4) << 3;          // 0 or 8
    const bf16_t* p = W + (size_t)n * ldw + kbase + ko;
    V16BF u;
    u.q[0] = *(const uint4*)(p);
    u.q[1] = *(const uint4*)(p + 16);
    return u.v;
}
__device__ inline v8f wmma_bf16(v16bf a, v16bf b, v8f c) {
    return __builtin_amdgcn_wmma_f32_16x16x32_bf16(false, a, false, b, (short)0, c, false, false);
}

__device__ inline float bflo(u32 u) { return __uint_as_float(u << 16); }
__device__ inline float bfhi(u32 u) { return __uint_as_float(u & 0xffff0000u); }

// ---------------- phase 0: weight conversion ----------------
__global__ void cvt_f32_to_bf16(const float* __restrict__ src, bf16_t* __restrict__ dst, int n) {
    int i = blockIdx.x * blockDim.x + threadIdx.x;
    if (i < n) dst[i] = (bf16_t)src[i];
}
// Wcat[n][k] : k<1024 -> W_ih[n][k] ; else W_hh[n][k-1024]   (2048 x 1536)
__global__ void build_wcat(const float* __restrict__ Wih, const float* __restrict__ Whh,
                           bf16_t* __restrict__ Wcat) {
    int i = blockIdx.x * blockDim.x + threadIdx.x;   // over 2048*1536
    if (i >= H4 * KX) return;
    int n = i / KX, k = i - n * KX;
    float v = (k < 2 * Hz) ? Wih[(size_t)n * (2 * Hz) + k] : Whh[(size_t)n * Hz + (k - 2 * Hz)];
    Wcat[i] = (bf16_t)v;
}

// ---------------- phase 0: keys/values GEMM (WMMA) ----------------
// one wave computes one 16-row M tile x 4 N-tiles, for BOTH keys and values
__global__ __launch_bounds__(256) void gemm_kv(const float* __restrict__ seq,
                                               const bf16_t* __restrict__ Wk,
                                               const bf16_t* __restrict__ Wv,
                                               const float* __restrict__ bk,
                                               const float* __restrict__ bv,
                                               bf16_t* __restrict__ keys,
                                               bf16_t* __restrict__ values) {
    int wid  = blockIdx.x * 8 + (threadIdx.x >> 5);   // 65536 waves
    int lane = threadIdx.x & 31;
    int mt   = wid >> 3;            // 0..8191  (16 rows of B*L each)
    int nq   = wid & 7;             // quad of N tiles (64 cols)
    const float* A = seq + (size_t)mt * 16 * Hz;

    V8F ak[4], av[4];
#pragma unroll
    for (int j = 0; j < 4; ++j) { ak[j].v = (v8f){}; av[j].v = (v8f){}; }

    for (int k = 0; k < Hz; k += 32) {
        v16bf a = load_a_f32(A, Hz, k, lane);
#pragma unroll
        for (int j = 0; j < 4; ++j) {
            int n0 = (nq * 4 + j) * 16;
            v16bf bkt = load_b_w(Wk, Hz, n0, k, lane);
            v16bf bvt = load_b_w(Wv, Hz, n0, k, lane);
            ak[j].v = wmma_bf16(a, bkt, ak[j].v);
            av[j].v = wmma_bf16(a, bvt, av[j].v);
        }
    }
    int rbase = (lane >> 4) << 3;       // 0 or 8
    int clane = lane & 15;
#pragma unroll
    for (int j = 0; j < 4; ++j) {
        int n0 = (nq * 4 + j) * 16;
        int col = n0 + clane;
        float bbk = bk[col], bbv = bv[col];
#pragma unroll
        for (int r = 0; r < 8; ++r) {
            size_t row = (size_t)mt * 16 + rbase + r;
            keys  [row * Hz + col] = (bf16_t)(ak[j].e[r] + bbk);
            values[row * Hz + col] = (bf16_t)(av[j].e[r] + bbv);
        }
    }
}

// ---------------- state init ----------------
__global__ void init_state(float* h, float* c, int* prev_y) {
    int i = blockIdx.x * blockDim.x + threadIdx.x;
    if (i < Bz * Hz) { h[i] = 0.f; c[i] = 0.f; }
    if (i < Bz) prev_y[i] = 0;   // SOS
}

// ---------------- per-step: build x = [emb(prev_y) | ctx | h] in bf16 ----------------
__global__ void build_xh(const int* __restrict__ prev_y, const float* __restrict__ E,
                         const float* __restrict__ ctx, const float* __restrict__ h,
                         bf16_t* __restrict__ xh) {
    int i = blockIdx.x * blockDim.x + threadIdx.x;   // over 64*1536
    if (i >= Bz * KX) return;
    int b = i / KX, j = i - b * KX;
    float v;
    if (j < Hz)            v = E[(size_t)prev_y[b] * Hz + j];
    else if (j < 2 * Hz)   v = ctx[(size_t)b * Hz + (j - Hz)];
    else                   v = h  [(size_t)b * Hz + (j - 2 * Hz)];
    xh[i] = (bf16_t)v;
}

// ---------------- per-step: LSTM gates GEMM (64x1536 @ 1536x2048) ----------------
__global__ __launch_bounds__(256) void lstm_gemm(const bf16_t* __restrict__ xh,
                                                 const bf16_t* __restrict__ Wcat,
                                                 const float* __restrict__ bih,
                                                 const float* __restrict__ bhh,
                                                 float* __restrict__ gates) {
    int wid  = blockIdx.x * 8 + (threadIdx.x >> 5);   // 512 waves
    int lane = threadIdx.x & 31;
    int mt   = wid >> 7;            // 0..3
    int nt   = wid & 127;           // 0..127
    const bf16_t* A = xh + (size_t)mt * 16 * KX;
    V8F acc; acc.v = (v8f){};
    for (int k = 0; k < KX; k += 32) {
        v16bf a = load_a_bf16(A, KX, k, lane);
        v16bf b = load_b_w(Wcat, KX, nt * 16, k, lane);
        acc.v = wmma_bf16(a, b, acc.v);
    }
    int rbase = (lane >> 4) << 3;
    int col   = nt * 16 + (lane & 15);
    float bias = bih[col] + bhh[col];
#pragma unroll
    for (int r = 0; r < 8; ++r) {
        int row = mt * 16 + rbase + r;
        gates[(size_t)row * H4 + col] = acc.e[r] + bias;
    }
}

// ---------------- per-step: LSTM pointwise ----------------
__device__ inline float sigf(float x) { return 1.0f / (1.0f + expf(-x)); }
__global__ void lstm_update(const float* __restrict__ gates, float* __restrict__ h,
                            float* __restrict__ c) {
    int i = blockIdx.x * blockDim.x + threadIdx.x;   // over 64*512
    if (i >= Bz * Hz) return;
    int b = i >> 9, k = i & 511;
    const float* g = gates + (size_t)b * H4;
    float ii = g[k], ff = g[Hz + k], gg = g[2 * Hz + k], oo = g[3 * Hz + k];
    float cn = sigf(ff) * c[i] + sigf(ii) * tanhf(gg);
    c[i] = cn;
    h[i] = sigf(oo) * tanhf(cn);
}

// ---------------- per-step: attention scores (masked) ----------------
__global__ __launch_bounds__(256) void att_scores(const bf16_t* __restrict__ keys,
                                                  const float* __restrict__ h,
                                                  const int* __restrict__ lens,
                                                  float* __restrict__ scores) {
    __shared__ float sh[Hz];
    int b = blockIdx.x >> 3;
    int l = (blockIdx.x & 7) * 256 + threadIdx.x;
    for (int j = threadIdx.x; j < Hz; j += 256) sh[j] = h[(size_t)b * Hz + j];
    __syncthreads();
    int len = lens[b];
    float sc = -3.4e38f;
    if (l < len) {
        const uint4* kp = (const uint4*)(keys + ((size_t)b * Lz + l) * Hz);
        float acc = 0.f;
        for (int it = 0; it < Hz / 8; ++it) {
            uint4 q = kp[it];
            int k8 = it * 8;
            acc += bflo(q.x) * sh[k8 + 0] + bfhi(q.x) * sh[k8 + 1];
            acc += bflo(q.y) * sh[k8 + 2] + bfhi(q.y) * sh[k8 + 3];
            acc += bflo(q.z) * sh[k8 + 4] + bfhi(q.z) * sh[k8 + 5];
            acc += bflo(q.w) * sh[k8 + 6] + bfhi(q.w) * sh[k8 + 7];
        }
        sc = acc * 0.044194173824159216f;   // 1/sqrt(512)
    }
    scores[(size_t)b * Lz + l] = sc;
}

// ---------------- per-step: softmax (in-place, one block per b) ----------------
__global__ __launch_bounds__(256) void att_softmax(float* __restrict__ scores) {
    __shared__ float red[256];
    int tid = threadIdx.x;
    float* s = scores + (size_t)blockIdx.x * Lz;
    float m = -3.4e38f;
    for (int j = tid; j < Lz; j += 256) m = fmaxf(m, s[j]);
    red[tid] = m; __syncthreads();
    for (int o = 128; o > 0; o >>= 1) { if (tid < o) red[tid] = fmaxf(red[tid], red[tid + o]); __syncthreads(); }
    m = red[0]; __syncthreads();
    float sum = 0.f;
    for (int j = tid; j < Lz; j += 256) { float e = expf(s[j] - m); s[j] = e; sum += e; }
    red[tid] = sum; __syncthreads();
    for (int o = 128; o > 0; o >>= 1) { if (tid < o) red[tid] += red[tid + o]; __syncthreads(); }
    float inv = 1.0f / red[0];
    for (int j = tid; j < Lz; j += 256) s[j] *= inv;
}

// ---------------- per-step: context = w @ values ----------------
__global__ __launch_bounds__(256) void att_ctx(const bf16_t* __restrict__ values,
                                               const float* __restrict__ w,
                                               float* __restrict__ ctx) {
    __shared__ float wl[256];
    int b = blockIdx.x, tid = threadIdx.x;
    const u32* vb = (const u32*)values + (size_t)b * Lz * (Hz / 2);
    float a0 = 0.f, a1 = 0.f;
    for (int lc = 0; lc < Lz / 256; ++lc) {
        wl[tid] = w[(size_t)b * Lz + lc * 256 + tid];
        __syncthreads();
        for (int li = 0; li < 256; ++li) {
            float wv = wl[li];                 // uniform across block
            if (wv != 0.0f) {                  // skip masked / zero-weight rows
                u32 u = vb[(size_t)(lc * 256 + li) * (Hz / 2) + tid];
                a0 += wv * bflo(u);
                a1 += wv * bfhi(u);
            }
        }
        __syncthreads();
    }
    ctx[(size_t)b * Hz + 2 * tid]     = a0;
    ctx[(size_t)b * Hz + 2 * tid + 1] = a1;
}

// ---------------- per-step: cdn logits + hash-Gumbel sample ----------------
__device__ inline u32 wang(u32 x) {
    x = (x ^ 61u) ^ (x >> 16); x *= 9u; x ^= x >> 4; x *= 0x27d4eb2du; x ^= x >> 15; return x;
}
__global__ __launch_bounds__(128) void cdn_sample(const float* __restrict__ h,
                                                  const float* __restrict__ ctx,
                                                  const float* __restrict__ Wcdn,
                                                  const float* __restrict__ bcdn,
                                                  int* __restrict__ prev_y,
                                                  int* __restrict__ out, int t) {
    __shared__ float hx[2 * Hz];
    __shared__ float lg[Vz];
    int b = blockIdx.x, tid = threadIdx.x;
    for (int j = tid; j < Hz; j += 128) {
        hx[j]      = h  [(size_t)b * Hz + j];
        hx[Hz + j] = ctx[(size_t)b * Hz + j];
    }
    __syncthreads();
    if (tid < Vz) {
        const float* wr = Wcdn + (size_t)tid * (2 * Hz);
        float acc = bcdn[tid];
        for (int k = 0; k < 2 * Hz; ++k) acc += hx[k] * wr[k];
        u32 hsh = wang((u32)t * 4096u + (u32)b * 64u + (u32)tid + 0x9E3779B9u);
        float u = (float)(hsh >> 8) * (1.0f / 16777216.0f) + 1e-7f;
        lg[tid] = acc - logf(-logf(u));     // Gumbel-max sampling
    }
    __syncthreads();
    if (tid == 0) {
        int best = 0; float bv = lg[0];
        for (int v = 1; v < Vz; ++v) if (lg[v] > bv) { bv = lg[v]; best = v; }
        prev_y[b] = best;
        out[(size_t)b * Tz + t] = best;
    }
}

// ---------------- launch ----------------
extern "C" void kernel_launch(void* const* d_in, const int* in_sizes, int n_in,
                              void* d_out, int out_size, void* d_ws, size_t ws_size,
                              hipStream_t stream) {
    const float* seq   = (const float*)d_in[0];
    const int*   lens  = (const int*)  d_in[1];
    const float* Echar = (const float*)d_in[2];
    const float* Wkey  = (const float*)d_in[3];
    const float* bkey  = (const float*)d_in[4];
    const float* Wval  = (const float*)d_in[5];
    const float* bval  = (const float*)d_in[6];
    const float* Wih   = (const float*)d_in[7];
    const float* Whh   = (const float*)d_in[8];
    const float* bih   = (const float*)d_in[9];
    const float* bhh   = (const float*)d_in[10];
    const float* Wcdn  = (const float*)d_in[11];
    const float* bcdn  = (const float*)d_in[12];
    int* out = (int*)d_out;

    char* ws = (char*)d_ws;
    size_t off = 0;
    bf16_t* keys   = (bf16_t*)(ws + off); off += (size_t)Bz * Lz * Hz * 2;   // 128 MB
    bf16_t* values = (bf16_t*)(ws + off); off += (size_t)Bz * Lz * Hz * 2;   // 128 MB
    bf16_t* Wk_b   = (bf16_t*)(ws + off); off += (size_t)Hz * Hz * 2;
    bf16_t* Wv_b   = (bf16_t*)(ws + off); off += (size_t)Hz * Hz * 2;
    bf16_t* Wcat   = (bf16_t*)(ws + off); off += (size_t)H4 * KX * 2;
    bf16_t* xh     = (bf16_t*)(ws + off); off += (size_t)Bz * KX * 2;
    float*  gates  = (float*) (ws + off); off += (size_t)Bz * H4 * 4;
    float*  hbuf   = (float*) (ws + off); off += (size_t)Bz * Hz * 4;
    float*  cbuf   = (float*) (ws + off); off += (size_t)Bz * Hz * 4;
    float*  ctx    = (float*) (ws + off); off += (size_t)Bz * Hz * 4;
    float*  scores = (float*) (ws + off); off += (size_t)Bz * Lz * 4;
    int*    prevy  = (int*)   (ws + off); off += 256;

    // ---- phase 0: convert weights, compute keys/values (WMMA), init state ----
    cvt_f32_to_bf16<<<(Hz * Hz + 255) / 256, 256, 0, stream>>>(Wkey, Wk_b, Hz * Hz);
    cvt_f32_to_bf16<<<(Hz * Hz + 255) / 256, 256, 0, stream>>>(Wval, Wv_b, Hz * Hz);
    build_wcat<<<(H4 * KX + 255) / 256, 256, 0, stream>>>(Wih, Whh, Wcat);
    gemm_kv<<<(Bz * Lz / 16) * 8 / 8, 256, 0, stream>>>(seq, Wk_b, Wv_b, bkey, bval, keys, values);
    init_state<<<(Bz * Hz + 255) / 256, 256, 0, stream>>>(hbuf, cbuf, prevy);
    // initial ctx with h = 0 (uniform attention over valid positions)
    att_scores<<<Bz * 8, 256, 0, stream>>>(keys, hbuf, lens, scores);
    att_softmax<<<Bz, 256, 0, stream>>>(scores);
    att_ctx<<<Bz, 256, 0, stream>>>(values, scores, ctx);

    // ---- decode loop ----
    for (int t = 0; t < Tz; ++t) {
        build_xh  <<<(Bz * KX + 255) / 256, 256, 0, stream>>>(prevy, Echar, ctx, hbuf, xh);
        lstm_gemm <<<64, 256, 0, stream>>>(xh, Wcat, bih, bhh, gates);
        lstm_update<<<(Bz * Hz + 255) / 256, 256, 0, stream>>>(gates, hbuf, cbuf);
        att_scores<<<Bz * 8, 256, 0, stream>>>(keys, hbuf, lens, scores);
        att_softmax<<<Bz, 256, 0, stream>>>(scores);
        att_ctx   <<<Bz, 256, 0, stream>>>(values, scores, ctx);
        cdn_sample<<<Bz, 128, 0, stream>>>(hbuf, ctx, Wcdn, bcdn, prevy, out, t);
    }
}
